// NTMBlock_30408368455698
// MI455X (gfx1250) — compile-verified, matrix operand order checked by reading, not run on hardware
//
#include <hip/hip_runtime.h>

// ---------------- problem constants ----------------
#define B_    32
#define T_    512
#define D_    512
#define H_    512
#define G4_   2048      // 4*H
#define N_    128
#define W_    128
#define IF_   524
#define KCAT  1024      // D + H
#define KOUT  640       // H + W
#define NKT_G 32        // K tiles (gates)  1024/32
#define NKT_I 16        // K tiles (iface)  512/32
#define NKT_O 20        // K tiles (out)    640/32
#define NT_I  33        // N tiles iface (528 padded cols)
#define NB    16        // persistent blocks
#define NTHR  256       // threads per block (8 waves)
#define EPS_  1e-8f

typedef unsigned short u16;
typedef __attribute__((ext_vector_type(16))) __bf16 v16bf;
typedef __attribute__((ext_vector_type(8)))  float  v8f;
typedef __attribute__((ext_vector_type(4)))  unsigned u32x4;
typedef __attribute__((ext_vector_type(4)))  int      i32x4;
typedef __attribute__((ext_vector_type(8)))  int      i32x8;

// ---------------- helpers ----------------
__device__ __forceinline__ u16 f2bf(float f) {
  unsigned u = __float_as_uint(f);
  unsigned r = (u + 0x7FFFu + ((u >> 16) & 1u)) >> 16;
  return (u16)r;
}

__device__ __forceinline__ float sigmoidf_(float x) { return 1.f / (1.f + __expf(-x)); }
__device__ __forceinline__ float softplusf_(float x) {
  return fmaxf(x, 0.f) + log1pf(__expf(-fabsf(x)));
}

// Offset (in u16 units) of element (rowcol, k_local) inside one 32-lane x 16-elem
// bf16 WMMA fragment, per CDNA5 ISA 7.12.2 16-bit A/B layout (wave32).
__device__ __forceinline__ int frag_off(int rc, int kl) {
  int base = kl & 15;
  int g = base >> 3;          // lane group
  int r = base & 7;
  int v = (kl < 16) ? (r >> 1) : (4 + (r >> 1));   // vgpr index 0..7
  int e = 2 * v + (r & 1);                          // element 0..15 within lane
  return ((rc + (g << 4)) << 4) + e;                // lane*16 + e
}

__device__ __forceinline__ v8f wmma_bf16(v16bf a, v16bf b, v8f c) {
  return __builtin_amdgcn_wmma_f32_16x16x32_bf16(false, a, false, b, (short)0, c,
                                                 false, false);
}

// ---- Tensor Data Mover: 1-D dword copy global -> LDS (CDNA5 ISA ch.8 D#) ----
// src/ldst must be 4-byte aligned; ndw <= 65535 dwords (tile_dim0 is 16-bit).
__device__ __forceinline__ void tdm_load_1d(const void* src, void* ldst, unsigned ndw) {
  unsigned long long ga = (unsigned long long)(uintptr_t)src;
  unsigned la = (unsigned)(uintptr_t)ldst;  // generic LDS addr: low 32 bits = LDS byte offset
  u32x4 g0;
  g0[0] = 1u;                                   // count=1 (valid), user mode, no gather
  g0[1] = la;                                   // lds_addr  (bits 63:32)
  g0[2] = (unsigned)ga;                         // global_addr[31:0]  (bits 95:64)
  g0[3] = ((unsigned)(ga >> 32) & 0x01FFFFFFu)  // global_addr[56:32]
          | (2u << 30);                         // type=2 ("image")   (bits 127:126)
  i32x8 g1;
  g1[0] = (int)(2u << 16);                      // workgroup_mask=0, data_size=2 (4 bytes)
  g1[1] = (int)((ndw & 0xFFFFu) << 16);         // tensor_dim0[15:0]
  g1[2] = (int)(((ndw >> 16) & 0xFFFFu) | (1u << 16)); // tensor_dim0[31:16], tensor_dim1=1
  g1[3] = (int)((ndw & 0xFFFFu) << 16);         // tile_dim0 = ndw
  g1[4] = 0;                                    // tile_dim1=0 (unused), tile_dim2=0
  g1[5] = (int)ndw;                             // tensor_dim0_stride lo
  g1[6] = 0;
  g1[7] = 0;                                    // tensor_dim1_stride
  i32x4 z4 = {0, 0, 0, 0};
#if defined(__clang_major__) && __clang_major__ >= 23
  i32x8 z8 = {0, 0, 0, 0, 0, 0, 0, 0};
  __builtin_amdgcn_tensor_load_to_lds(g0, g1, z4, z4, z8, 0);
#else
  __builtin_amdgcn_tensor_load_to_lds(g0, g1, z4, z4, 0);
#endif
}

// device-wide barrier (persistent kernel, NB blocks all resident)
__device__ __forceinline__ void grid_sync(unsigned* bar, int tid) {
  __threadfence();
  __syncthreads();
  if (tid == 0) {
    unsigned gen = __hip_atomic_load(&bar[1], __ATOMIC_RELAXED, __HIP_MEMORY_SCOPE_AGENT);
    unsigned arrived =
        __hip_atomic_fetch_add(&bar[0], 1u, __ATOMIC_ACQ_REL, __HIP_MEMORY_SCOPE_AGENT);
    if (arrived == NB - 1) {
      __hip_atomic_store(&bar[0], 0u, __ATOMIC_RELAXED, __HIP_MEMORY_SCOPE_AGENT);
      __hip_atomic_fetch_add(&bar[1], 1u, __ATOMIC_RELEASE, __HIP_MEMORY_SCOPE_AGENT);
    } else {
      while (__hip_atomic_load(&bar[1], __ATOMIC_ACQUIRE, __HIP_MEMORY_SCOPE_AGENT) == gen) {
        __builtin_amdgcn_s_sleep(1);
      }
    }
  }
  __syncthreads();
  __threadfence();
}

__device__ __forceinline__ float red_sum128(float v, float* s, int tid) {
  __syncthreads();
  if (tid < 128) s[tid] = v;
  __syncthreads();
  for (int off = 64; off >= 1; off >>= 1) {
    if (tid < off) s[tid] += s[tid + off];
    __syncthreads();
  }
  float r = s[0];
  __syncthreads();
  return r;
}

__device__ __forceinline__ float red_max128(float v, float* s, int tid) {
  __syncthreads();
  if (tid < 128) s[tid] = v;
  __syncthreads();
  for (int off = 64; off >= 1; off >>= 1) {
    if (tid < off) s[tid] = fmaxf(s[tid], s[tid + off]);
    __syncthreads();
  }
  float r = s[0];
  __syncthreads();
  return r;
}

// ---------------- prep kernels (run once per launch, parallel) ----------------
__global__ void k_prep(const float* bih, const float* bhh, const float* M0,
                       float* bias_g, float* cst, u16* hbf_sw, float* wr, float* wwt,
                       float* Mst, unsigned* bar) {
  int g = blockIdx.x * blockDim.x + threadIdx.x;
  int stride = gridDim.x * blockDim.x;
  for (int i = g; i < B_ * N_ * W_; i += stride) {
    Mst[i] = M0[i & (N_ * W_ - 1)];
    if (i < B_ * H_) { cst[i] = 0.f; hbf_sw[i] = 0; }
    if (i < B_ * N_) { wr[i] = 1.f / N_; wwt[i] = 1.f / N_; }
    if (i < G4_) bias_g[i] = bih[i] + bhh[i];
    if (i < 2) bar[i] = 0u;
  }
}

// pre-swizzle x into per-timestep A-fragment layout (32KB contiguous per t)
__global__ void k_sw_x(const float* x, u16* xsw) {
  int g = blockIdx.x * blockDim.x + threadIdx.x;
  int stride = gridDim.x * blockDim.x;
  for (int i = g; i < B_ * T_ * D_; i += stride) {
    int k = i & 511;
    int mt = i >> 9;
    int t = mt & 511;
    int m = mt >> 9;
    xsw[(size_t)t * (B_ * 512) + ((m >> 4) * 16 + (k >> 5)) * 512 +
        frag_off(m & 15, k & 31)] = f2bf(x[i]);
  }
}

// swizzle [Wih|Whh]^T into B-fragment layout: tile (nt,kt) is 32 lanes x 16 bf16
__global__ void k_sw_wcat(const float* Wih, const float* Whh, u16* dst) {
  int g = blockIdx.x * blockDim.x + threadIdx.x;
  int stride = gridDim.x * blockDim.x;
  for (int i = g; i < G4_ * KCAT; i += stride) {
    int j = i >> 10, k = i & (KCAT - 1);
    float v = (k < D_) ? Wih[j * D_ + k] : Whh[j * H_ + (k - D_)];
    dst[((j >> 4) * NKT_G + (k >> 5)) * 512 + frag_off(j & 15, k & 31)] = f2bf(v);
  }
}

__global__ void k_sw_iface(const float* W, u16* dst) {
  int g = blockIdx.x * blockDim.x + threadIdx.x;
  int stride = gridDim.x * blockDim.x;
  for (int i = g; i < 528 * 512; i += stride) {
    int n = i >> 9, k = i & 511;
    float v = (n < IF_) ? W[n * 512 + k] : 0.f;
    dst[((n >> 4) * NKT_I + (k >> 5)) * 512 + frag_off(n & 15, k & 31)] = f2bf(v);
  }
}

__global__ void k_sw_out(const float* W, u16* dst) {
  int g = blockIdx.x * blockDim.x + threadIdx.x;
  int stride = gridDim.x * blockDim.x;
  for (int i = g; i < D_ * KOUT; i += stride) {
    int n = i / KOUT, k = i - n * KOUT;
    float v = W[n * KOUT + k];
    dst[((n >> 4) * NKT_O + (k >> 5)) * 512 + frag_off(n & 15, k & 31)] = f2bf(v);
  }
}

// ---------------- the persistent NTM kernel ----------------
__global__ void __launch_bounds__(NTHR) k_ntm(
    const u16* __restrict__ xsw, u16* __restrict__ hbf_sw, float* __restrict__ cst,
    float* __restrict__ gates, float* __restrict__ pbuf, float* __restrict__ Mst,
    float* __restrict__ wr, float* __restrict__ wwt, float* __restrict__ rt,
    const u16* __restrict__ wcatB, const u16* __restrict__ ifaceB,
    const u16* __restrict__ outB, const float* __restrict__ bias_g,
    const float* __restrict__ iface_b, const float* __restrict__ out_b,
    float* __restrict__ out, unsigned* bar) {
  __shared__ __attribute__((aligned(32))) u16 sA[2 * NKT_O * 512];  // 40KB staging
  __shared__ float sred[128];
  __shared__ float stmp[128];
  __shared__ float swr[128];
  __shared__ float sww[128];

  const int tid = threadIdx.x;
  const int lane = tid & 31;
  const int wv = blockIdx.x * (NTHR / 32) + (tid >> 5);  // global wave 0..127

  for (int t = 0; t < T_; ++t) {
    // ================= Phase A: gates = [x_t|h] @ Wcat^T + b  (32x1024x2048)
    v8f acc0 = {0.f, 0.f, 0.f, 0.f, 0.f, 0.f, 0.f, 0.f};
    v8f acc1 = {0.f, 0.f, 0.f, 0.f, 0.f, 0.f, 0.f, 0.f};
    for (int ch = 0; ch < 2; ++ch) {  // K in two 512 chunks (x then h)
      if (tid < 32) {  // TDM: one DMA of the pre-swizzled 32KB A panel into LDS
        const void* src = (ch == 0) ? (const void*)(xsw + (size_t)t * (B_ * 512))
                                    : (const void*)hbf_sw;
        tdm_load_1d(src, sA, (B_ * 512 * 2) / 4);
        __builtin_amdgcn_s_wait_tensorcnt(0);
      }
      __syncthreads();
      const u16* bb = wcatB + (size_t)wv * (NKT_G * 512) + (size_t)ch * 16 * 512;
#pragma unroll 4
      for (int kt = 0; kt < 16; ++kt) {
        __builtin_prefetch(bb + (kt + 4) * 512 + lane * 16, 0, 1);
        v16bf bf = *(const v16bf*)(bb + kt * 512 + lane * 16);
        v16bf a0 = *(const v16bf*)(sA + kt * 512 + lane * 16);
        v16bf a1 = *(const v16bf*)(sA + (16 + kt) * 512 + lane * 16);
        acc0 = wmma_bf16(a0, bf, acc0);
        acc1 = wmma_bf16(a1, bf, acc1);
      }
      __syncthreads();
    }
    {
      int col = wv * 16 + (lane & 15);
      int r0 = (lane >> 4) * 8;
      float bg = bias_g[col];
#pragma unroll
      for (int v2 = 0; v2 < 8; ++v2) {
        gates[(r0 + v2) * G4_ + col] = acc0[v2] + bg;
        gates[(16 + r0 + v2) * G4_ + col] = acc1[v2] + bg;
      }
    }
    grid_sync(bar, tid);

    // ================= Phase B: LSTM pointwise; h -> bf16 (fragment-swizzled)
    {
      int gtid = blockIdx.x * NTHR + tid;
      for (int i = gtid; i < B_ * H_; i += NB * NTHR) {
        int b2 = i >> 9, j = i & 511;
        const float* gr = gates + b2 * G4_;
        float ig = sigmoidf_(gr[j]);
        float fg = sigmoidf_(gr[512 + j]);
        float g2 = tanhf(gr[1024 + j]);
        float og = sigmoidf_(gr[1536 + j]);
        float cn = fg * cst[i] + ig * g2;
        cst[i] = cn;
        hbf_sw[((b2 >> 4) * 16 + (j >> 5)) * 512 + frag_off(b2 & 15, j & 31)] =
            f2bf(og * tanhf(cn));
      }
    }
    grid_sync(bar, tid);

    // ================= Phase C: p = h @ iface_W^T + iface_b  (32x512x524)
    if (blockIdx.x < 5) {
      if (tid < 32) {
        tdm_load_1d(hbf_sw, sA, (B_ * 512 * 2) / 4);
        __builtin_amdgcn_s_wait_tensorcnt(0);
      }
      __syncthreads();
      if (wv < NT_I) {
        v8f p0 = {0.f, 0.f, 0.f, 0.f, 0.f, 0.f, 0.f, 0.f};
        v8f p1 = {0.f, 0.f, 0.f, 0.f, 0.f, 0.f, 0.f, 0.f};
        const u16* bb = ifaceB + (size_t)wv * (NKT_I * 512);
#pragma unroll 4
        for (int kt = 0; kt < NKT_I; ++kt) {
          v16bf bf = *(const v16bf*)(bb + kt * 512 + lane * 16);
          v16bf a0 = *(const v16bf*)(sA + kt * 512 + lane * 16);
          v16bf a1 = *(const v16bf*)(sA + (NKT_I + kt) * 512 + lane * 16);
          p0 = wmma_bf16(a0, bf, p0);
          p1 = wmma_bf16(a1, bf, p1);
        }
        int col = wv * 16 + (lane & 15);
        int r0 = (lane >> 4) * 8;
        if (col < IF_) {
          float ib = iface_b[col];
#pragma unroll
          for (int v2 = 0; v2 < 8; ++v2) {
            pbuf[(r0 + v2) * IF_ + col] = p0[v2] + ib;
            pbuf[(16 + r0 + v2) * IF_ + col] = p1[v2] + ib;
          }
        }
      }
    }
    grid_sync(bar, tid);

    // ================= Phase D: addressing + read + memory write (2 batches/block)
    for (int bi = 0; bi < 2; ++bi) {
      int b = blockIdx.x * 2 + bi;
      const float* pb = pbuf + b * IF_;
      float* Mb = Mst + (size_t)b * (N_ * W_);
      for (int head = 0; head < 2; ++head) {
        int koff = head == 0 ? 0 : 134;
        int boff = head == 0 ? 128 : 262;
        int goff = head == 0 ? 129 : 263;
        int soff = head == 0 ? 130 : 264;
        int gmoff = head == 0 ? 133 : 267;
        float* wprev = (head == 0 ? wr : wwt) + b * N_;
        float beta = softplusf_(pb[boff]);
        float gg = sigmoidf_(pb[goff]);
        float s0r = pb[soff], s1r = pb[soff + 1], s2r = pb[soff + 2];
        float smx = fmaxf(s0r, fmaxf(s1r, s2r));
        float e0 = __expf(s0r - smx), e1 = __expf(s1r - smx), e2 = __expf(s2r - smx);
        float esn = e0 + e1 + e2;
        float sh0 = e0 / esn, sh1 = e1 / esn, sh2 = e2 / esn;
        float gamma = fmaxf(1.f + softplusf_(pb[gmoff]), 1.f);

        float dot = 0.f, nrm = 0.f, kk = 0.f;
        if (tid < N_) {
          const float* Mrow = Mb + tid * W_;
          float kv = pb[koff + tid];
          kk = kv * kv;
          for (int w2 = 0; w2 < W_; ++w2) {
            float m = Mrow[w2];
            dot += pb[koff + w2] * m;
            nrm += m * m;
          }
        }
        float sumk2 = red_sum128(kk, sred, tid);
        float sim = dot * rsqrtf(sumk2 + EPS_) * rsqrtf(nrm + EPS_);
        float a = beta * sim;
        float amax = red_max128(tid < N_ ? a : -1e30f, sred, tid);
        float ev = (tid < N_) ? __expf(a - amax) : 0.f;
        float es2 = red_sum128(ev, sred, tid);
        float wc = ev / es2;
        float wi = gg * wc + (1.f - gg) * ((tid < N_) ? wprev[tid] : 0.f);
        if (tid < N_) stmp[tid] = wi;
        __syncthreads();
        float wsft = 0.f;
        if (tid < N_)
          wsft = sh0 * stmp[(tid + 1) & 127] + sh1 * stmp[tid] + sh2 * stmp[(tid - 1) & 127];
        __syncthreads();
        float wpow = (tid < N_) ? __powf(fmaxf(wsft, EPS_), gamma) : 0.f;
        float wsum = red_sum128(wpow, sred, tid);
        float wfin = wpow / (wsum + EPS_);
        if (tid < N_) {
          (head == 0 ? swr : sww)[tid] = wfin;
          wprev[tid] = wfin;
        }
        __syncthreads();
      }
      // read vector r_t (pre-update M), coalesced over w
      if (tid < W_) {
        float acc = 0.f;
        for (int n = 0; n < N_; ++n) acc += swr[n] * Mb[n * W_ + tid];
        rt[b * W_ + tid] = acc;
      }
      __syncthreads();
      // memory write: M = M*(1 - w_w*erase) + w_w*add
      for (int idx = tid; idx < N_ * W_; idx += NTHR) {
        int n = idx >> 7, w2 = idx & 127;
        float e = sigmoidf_(pb[268 + w2]);
        float ad = tanhf(pb[396 + w2]);
        float wwv = sww[n];
        Mb[idx] = Mb[idx] * (1.f - wwv * e) + wwv * ad;
      }
      __syncthreads();
    }
    grid_sync(bar, tid);

    // ================= Phase E: y = [h|r] @ out_W^T + out_b  (32x640x512)
    if (blockIdx.x < 4) {
      if (tid < 32) {  // TDM: copy the two 16KB h tile-blocks into NKT_O-tiled slots
        tdm_load_1d(hbf_sw, sA, (16 * 512 * 2) / 4);
        tdm_load_1d(hbf_sw + 16 * 512, sA + NKT_O * 512, (16 * 512 * 2) / 4);
        __builtin_amdgcn_s_wait_tensorcnt(0);
      }
      // r-part (tiles kt=16..19 of each m-tile), disjoint from TDM destination
      for (int i = tid; i < B_ * W_; i += NTHR) {
        int m = i >> 7, kw = i & 127;
        sA[((m >> 4) * NKT_O + 16 + (kw >> 5)) * 512 + frag_off(m & 15, kw & 31)] =
            f2bf(rt[i]);
      }
      __syncthreads();
      v8f y0 = {0.f, 0.f, 0.f, 0.f, 0.f, 0.f, 0.f, 0.f};
      v8f y1 = {0.f, 0.f, 0.f, 0.f, 0.f, 0.f, 0.f, 0.f};
      const u16* bb = outB + (size_t)wv * (NKT_O * 512);
#pragma unroll 4
      for (int kt = 0; kt < NKT_O; ++kt) {
        v16bf bf = *(const v16bf*)(bb + kt * 512 + lane * 16);
        v16bf a0 = *(const v16bf*)(sA + kt * 512 + lane * 16);
        v16bf a1 = *(const v16bf*)(sA + (NKT_O + kt) * 512 + lane * 16);
        y0 = wmma_bf16(a0, bf, y0);
        y1 = wmma_bf16(a1, bf, y1);
      }
      int col = wv * 16 + (lane & 15);
      int r0 = (lane >> 4) * 8;
      float ob = out_b[col];
#pragma unroll
      for (int v2 = 0; v2 < 8; ++v2) {
        out[(size_t)(r0 + v2) * (T_ * D_) + (size_t)t * D_ + col] = y0[v2] + ob;
        out[(size_t)(16 + r0 + v2) * (T_ * D_) + (size_t)t * D_ + col] = y1[v2] + ob;
      }
      __syncthreads();  // protect sA before next-step staging
    }
    // no grid barrier needed here: next-step barriers cover remaining hazards
  }
}

// ---------------- launcher ----------------
extern "C" void kernel_launch(void* const* d_in, const int* in_sizes, int n_in,
                              void* d_out, int out_size, void* d_ws, size_t ws_size,
                              hipStream_t stream) {
  (void)in_sizes; (void)n_in; (void)out_size; (void)ws_size;
  const float* x       = (const float*)d_in[0];
  const float* Wih     = (const float*)d_in[1];
  const float* Whh     = (const float*)d_in[2];
  const float* bih     = (const float*)d_in[3];
  const float* bhh     = (const float*)d_in[4];
  const float* iface_W = (const float*)d_in[5];
  const float* iface_b = (const float*)d_in[6];
  const float* out_W   = (const float*)d_in[7];
  const float* out_b   = (const float*)d_in[8];
  const float* M0      = (const float*)d_in[9];
  float* out = (float*)d_out;

  char* ws = (char*)d_ws;
  size_t off = 0;
  auto take = [&](size_t bytes) { char* p = ws + off; off += (bytes + 255) & ~(size_t)255; return p; };
  unsigned* bar   = (unsigned*)take(64);
  u16* wcatB      = (u16*)take((size_t)G4_ * KCAT * 2);
  u16* ifaceB     = (u16*)take((size_t)528 * 512 * 2);
  u16* outB       = (u16*)take((size_t)D_ * KOUT * 2);
  float* bias_g   = (float*)take((size_t)G4_ * 4);
  u16* xsw        = (u16*)take((size_t)B_ * T_ * D_ * 2);
  u16* hbf_sw     = (u16*)take((size_t)B_ * H_ * 2);
  float* cst      = (float*)take((size_t)B_ * H_ * 4);
  float* gates    = (float*)take((size_t)B_ * G4_ * 4);
  float* pbuf     = (float*)take((size_t)B_ * IF_ * 4);
  float* Mst      = (float*)take((size_t)B_ * N_ * W_ * 4);
  float* wr       = (float*)take((size_t)B_ * N_ * 4);
  float* wwt      = (float*)take((size_t)B_ * N_ * 4);
  float* rt       = (float*)take((size_t)B_ * W_ * 4);

  k_prep<<<256, 256, 0, stream>>>(bih, bhh, M0, bias_g, cst, hbf_sw, wr, wwt, Mst, bar);
  k_sw_x<<<1024, 256, 0, stream>>>(x, xsw);
  k_sw_wcat<<<512, 256, 0, stream>>>(Wih, Whh, wcatB);
  k_sw_iface<<<128, 256, 0, stream>>>(iface_W, ifaceB);
  k_sw_out<<<160, 256, 0, stream>>>(out_W, outB);
  k_ntm<<<NB, NTHR, 0, stream>>>(xsw, hbf_sw, cst, gates, pbuf, Mst, wr, wwt, rt,
                                 wcatB, ifaceB, outB, bias_g, iface_b, out_b, out, bar);
}
